// MyOperator_56710748176848
// MI455X (gfx1250) — matrix operand order repaired and checked by reading, not compiled, not run on hardware
//
#include <hip/hip_runtime.h>
#include <hip/hip_bf16.h>

// ---------------------------------------------------------------------------
// MoE-MLP chain for MI455X (gfx1250), wave32 + WMMA f32_16x16x32_f16.
//   L=8 layers, E=8 experts, D=256, H1=64, H2=32, B=16384.
// Everything fits in L2 (weights 6.5MB, x 16MB) -> compute bound on WMMA.
// Weights converted fp32->f16 once into d_ws; activations kept f16 in LDS,
// f32 accumulation. Softmax weight folded into h2 before GEMM3 so all
// experts accumulate into one WMMA accumulator set.
// ---------------------------------------------------------------------------

typedef __attribute__((ext_vector_type(16))) _Float16 v16h;
typedef __attribute__((ext_vector_type(8)))  _Float16 v8h_t;
typedef __attribute__((ext_vector_type(8)))  float    v8f;

static constexpr int kL  = 8;
static constexpr int kE  = 8;
static constexpr int kD  = 256;
static constexpr int kB  = 16384;
static constexpr int kH1 = 64;
static constexpr int kH2 = 32;
static constexpr int kRowsPerBlock = 64;   // 4 waves x 16 rows

// ---- fragment loaders -------------------------------------------------------
// A (16x32 f16) from row-major LDS tile: lane = hlf*16 + m16.
//   e=0..15 -> k = k0 + hlf*16 + e       (32B contiguous)
__device__ __forceinline__ v16h load_a_lds(const _Float16* base, int ldm,
                                           int k0, int m16, int hlf) {
  const _Float16* p = base + m16 * ldm + k0 + hlf * 16;
  v8h_t lo = *reinterpret_cast<const v8h_t*>(p);
  v8h_t hi = *reinterpret_cast<const v8h_t*>(p + 8);
  return __builtin_shufflevector(lo, hi, 0,1,2,3,4,5,6,7,8,9,10,11,12,13,14,15);
}

// B (32x16 f16) from row-major [N][K] f16 weights in global:
// lane holds column n, 16 contiguous K values starting at k (caller folds
// hlf*16 into k). 2x 16B loads.
__device__ __forceinline__ v16h load_b_glb(const _Float16* __restrict__ W,
                                           int ldk, int n, int k) {
  const _Float16* p = W + (size_t)n * ldk + k;
  v8h_t lo = *reinterpret_cast<const v8h_t*>(p);
  v8h_t hi = *reinterpret_cast<const v8h_t*>(p + 8);
  return __builtin_shufflevector(lo, hi, 0,1,2,3,4,5,6,7,8,9,10,11,12,13,14,15);
}

__device__ __forceinline__ v8f wmma16(v16h a, v16h b, v8f c) {
  return __builtin_amdgcn_wmma_f32_16x16x32_f16(
      /*neg_a=*/false, a, /*neg_b=*/false, b,
      /*c_mod=*/(short)0, c, /*reuse_a=*/false, /*reuse_b=*/false);
}

// ---- prep: fp32 -> f16 weight conversion -----------------------------------
__global__ void cvt_f32_to_f16(const float* __restrict__ src,
                               _Float16* __restrict__ dst, int n) {
  int i = blockIdx.x * blockDim.x + threadIdx.x;
  int stride = gridDim.x * blockDim.x;
  for (; i < n; i += stride) dst[i] = (_Float16)src[i];
}

// ---- main fused kernel ------------------------------------------------------
__global__ __launch_bounds__(128) void moe_chain_kernel(
    const float*   __restrict__ x0,     // [B][D]
    const _Float16* __restrict__ w1h,   // [L][E][H1][D]
    const _Float16* __restrict__ w2h,   // [L][E][H2][H1]
    const _Float16* __restrict__ w3h,   // [L][E][D][H2]
    const float*   __restrict__ b1,     // [L][E][H1]
    const float*   __restrict__ b2,     // [L][E][H2]
    const float*   __restrict__ b3,     // [L][E][D]
    const float*   __restrict__ masks,  // [L][B][E]
    float*         __restrict__ out)    // [B][D]
{
  __shared__ __align__(16) _Float16 xs[kRowsPerBlock][264]; // 256 + 8 pad
  __shared__ __align__(16) _Float16 h1s[4][16][72];         // 64 + 8 pad
  __shared__ __align__(16) _Float16 h2s[4][16][40];         // 32 + 8 pad
  __shared__ float wsm[4][16][kE];                          // softmax weights

  const int lane = threadIdx.x & 31;
  const int wave = threadIdx.x >> 5;
  const int m16  = lane & 15;
  const int hlf  = lane >> 4;
  const int wr   = wave * 16;
  const int rowbase = blockIdx.x * kRowsPerBlock + wr;

  // ---- stage this wave's 16 rows of x as f16 in LDS (lanes split a row) ----
  for (int r = 0; r < 16; ++r) {
    const float4* src = reinterpret_cast<const float4*>(
        x0 + (size_t)(rowbase + r) * kD);
    float4 a = src[lane * 2 + 0];
    float4 b = src[lane * 2 + 1];
    _Float16* dst = &xs[wr + r][lane * 8];
    dst[0] = (_Float16)a.x; dst[1] = (_Float16)a.y;
    dst[2] = (_Float16)a.z; dst[3] = (_Float16)a.w;
    dst[4] = (_Float16)b.x; dst[5] = (_Float16)b.y;
    dst[6] = (_Float16)b.z; dst[7] = (_Float16)b.w;
  }

  for (int l = 0; l < kL; ++l) {
    const _Float16* W1l = w1h + (size_t)l * kE * kH1 * kD;
    const _Float16* W2l = w2h + (size_t)l * kE * kH2 * kH1;
    const _Float16* W3l = w3h + (size_t)l * kE * kD * kH2;
    const float* b1l = b1 + (size_t)l * kE * kH1;
    const float* b2l = b2 + (size_t)l * kE * kH2;
    const float* b3l = b3 + (size_t)l * kE * kD;

    // ---- softmax(mask) over experts for this wave's 16 rows --------------
    if (lane < 16) {
      const float* mp = masks + ((size_t)l * kB + rowbase + lane) * kE;
      float mv[kE];
      float mx = mp[0];
#pragma unroll
      for (int e = 0; e < kE; ++e) { mv[e] = mp[e]; mx = fmaxf(mx, mv[e]); }
      float s = 0.f;
#pragma unroll
      for (int e = 0; e < kE; ++e) { mv[e] = __expf(mv[e] - mx); s += mv[e]; }
      float inv = 1.0f / s;
#pragma unroll
      for (int e = 0; e < kE; ++e) wsm[wave][lane][e] = mv[e] * inv;
    }

    // x_new accumulators: 16 n-tiles of D, f32
    v8f acc[16];
#pragma unroll
    for (int nt = 0; nt < 16; ++nt)
      acc[nt] = (v8f){0.f,0.f,0.f,0.f,0.f,0.f,0.f,0.f};

    for (int e = 0; e < kE; ++e) {
      // softmax weight for rows (r + hlf*8) handled by this lane's C slots
      float wv[8];
#pragma unroll
      for (int r = 0; r < 8; ++r) wv[r] = wsm[wave][r + hlf * 8][e];

      const _Float16* W1e = W1l + (size_t)e * kH1 * kD;
      const _Float16* W2e = W2l + (size_t)e * kH2 * kH1;
      const _Float16* W3e = W3l + (size_t)e * kD * kH2;

      // ---- GEMM1: h1 = relu(x @ W1^T + b1)  [16 x 64], K = 256 -----------
#pragma unroll
      for (int nt = 0; nt < 4; ++nt) {
        v8f c = (v8f){0.f,0.f,0.f,0.f,0.f,0.f,0.f,0.f};
#pragma unroll
        for (int kt = 0; kt < 8; ++kt) {
          v16h a = load_a_lds(&xs[wr][0], 264, kt * 32, m16, hlf);
          v16h b = load_b_glb(W1e, kD, nt * 16 + m16, kt * 32 + hlf * 16);
          c = wmma16(a, b, c);
        }
        float bias = b1l[e * kH1 + nt * 16 + m16];
#pragma unroll
        for (int r = 0; r < 8; ++r) {
          float v = c[r] + bias;
          v = v > 0.f ? v : 0.f;
          h1s[wave][r + hlf * 8][nt * 16 + m16] = (_Float16)v;
        }
      }

      // ---- GEMM2: h2 = relu(h1 @ W2^T + b2) * w  [16 x 32], K = 64 -------
#pragma unroll
      for (int nt = 0; nt < 2; ++nt) {
        v8f c = (v8f){0.f,0.f,0.f,0.f,0.f,0.f,0.f,0.f};
#pragma unroll
        for (int kt = 0; kt < 2; ++kt) {
          v16h a = load_a_lds(&h1s[wave][0][0], 72, kt * 32, m16, hlf);
          v16h b = load_b_glb(W2e, kH1, nt * 16 + m16, kt * 32 + hlf * 16);
          c = wmma16(a, b, c);
        }
        float bias = b2l[e * kH2 + nt * 16 + m16];
#pragma unroll
        for (int r = 0; r < 8; ++r) {
          float v = c[r] + bias;
          v = v > 0.f ? v : 0.f;
          h2s[wave][r + hlf * 8][nt * 16 + m16] = (_Float16)(v * wv[r]);
        }
      }

      // ---- GEMM3: acc += (w.h2) @ W3^T  [16 x 256], K = 32 (1 step) ------
      v16h a3 = load_a_lds(&h2s[wave][0][0], 40, 0, m16, hlf);
#pragma unroll
      for (int nt = 0; nt < 16; ++nt) {
        v16h b = load_b_glb(W3e, kH2, nt * 16 + m16, hlf * 16);
        acc[nt] = wmma16(a3, b, acc[nt]);
        float b3v = b3l[e * kD + nt * 16 + m16];
#pragma unroll
        for (int r = 0; r < 8; ++r) acc[nt][r] += wv[r] * b3v;
      }
    } // experts

    // ---- write x_new: back to LDS (f16) or final output (f32) -------------
    if (l < kL - 1) {
#pragma unroll
      for (int nt = 0; nt < 16; ++nt)
#pragma unroll
        for (int r = 0; r < 8; ++r)
          xs[wr + r + hlf * 8][nt * 16 + m16] = (_Float16)acc[nt][r];
    } else {
#pragma unroll
      for (int nt = 0; nt < 16; ++nt)
#pragma unroll
        for (int r = 0; r < 8; ++r)
          out[(size_t)(rowbase + r + hlf * 8) * kD + nt * 16 + m16] =
              acc[nt][r];
    }
  } // layers
}

// ---------------------------------------------------------------------------
extern "C" void kernel_launch(void* const* d_in, const int* in_sizes, int n_in,
                              void* d_out, int out_size, void* d_ws,
                              size_t ws_size, hipStream_t stream) {
  (void)in_sizes; (void)n_in; (void)out_size; (void)ws_size;
  const float* x0    = (const float*)d_in[0];  // src_embeddings [B,D]
  const float* W1    = (const float*)d_in[1];  // [L,E,H1,D]
  const float* b1    = (const float*)d_in[2];  // [L,E,H1]
  const float* W2    = (const float*)d_in[3];  // [L,E,H2,H1]
  const float* b2    = (const float*)d_in[4];  // [L,E,H2]
  const float* W3    = (const float*)d_in[5];  // [L,E,D,H2]
  const float* b3    = (const float*)d_in[6];  // [L,E,D]
  const float* masks = (const float*)d_in[7];  // [L,B,E]
  float* out = (float*)d_out;

  const int n1 = kL * kE * kH1 * kD;   // 1,048,576
  const int n2 = kL * kE * kH2 * kH1;  //   131,072
  const int n3 = kL * kE * kD * kH2;   // 524,288

  _Float16* w1h = (_Float16*)d_ws;     // ~3.4 MB of ws used, 16B-aligned cuts
  _Float16* w2h = w1h + (size_t)n1;
  _Float16* w3h = w2h + (size_t)n2;

  cvt_f32_to_f16<<<(n1 + 255) / 256, 256, 0, stream>>>(W1, w1h, n1);
  cvt_f32_to_f16<<<(n2 + 255) / 256, 256, 0, stream>>>(W2, w2h, n2);
  cvt_f32_to_f16<<<(n3 + 255) / 256, 256, 0, stream>>>(W3, w3h, n3);

  moe_chain_kernel<<<kB / kRowsPerBlock, 128, 0, stream>>>(
      x0, w1h, w2h, w3h, b1, b2, b3, masks, out);
}
